// MultiHeadAttention_61452392071226
// MI455X (gfx1250) — matrix operand order repaired and checked by reading, not compiled, not run on hardware
//
#include <hip/hip_runtime.h>
#include <hip/hip_bf16.h>

// ---------------------------------------------------------------------------
// Types for the CDNA5 WMMA ABI (wave32, v_wmma_f32_16x16x32_bf16)
// ---------------------------------------------------------------------------
typedef __attribute__((ext_vector_type(16))) __bf16 bf16x16;
typedef __attribute__((ext_vector_type(8)))  float  f32x8;
typedef __attribute__((ext_vector_type(4)))  int    v4i;

struct alignas(16) U4 { unsigned int x, y, z, w; };
struct U8 { U4 a, b; };

__device__ __forceinline__ unsigned short f2bf(float f) {
    unsigned int u = __float_as_uint(f);
    u += 0x7FFFu + ((u >> 16) & 1u);          // round-to-nearest-even
    return (unsigned short)(u >> 16);
}
__device__ __forceinline__ float bf2f(unsigned short h) {
    return __uint_as_float(((unsigned int)h) << 16);
}
// Load one 16x32(A) / 32x16(B) bf16 fragment half-pair from LDS (two b128s).
__device__ __forceinline__ bf16x16 frag_ld(const unsigned short* p0,
                                           const unsigned short* p1) {
    U8 t;
    t.a = *(const U4*)p0;
    t.b = *(const U4*)p1;
    return __builtin_bit_cast(bf16x16, t);
}
__device__ __forceinline__ U4 pack8(const unsigned short* t) {
    U4 r;
    r.x = (unsigned)t[0] | ((unsigned)t[1] << 16);
    r.y = (unsigned)t[2] | ((unsigned)t[3] << 16);
    r.z = (unsigned)t[4] | ((unsigned)t[5] << 16);
    r.w = (unsigned)t[6] | ((unsigned)t[7] << 16);
    return r;
}

#define WMMA_BF16(A, B, C) \
    __builtin_amdgcn_wmma_f32_16x16x32_bf16(false, (A), false, (B), (short)0, (C), false, false)

// ---------------------------------------------------------------------------
// CDNA5 async global->LDS copy (GLOBAL_LOAD_ASYNC_TO_LDS_B128, ASYNCcnt),
// with a synchronous b128 fallback if the builtin is absent on this toolchain.
// Probed signature: (v4i addrspace(1)*, v4i addrspace(3)*, imm offset, imm cpol)
// ---------------------------------------------------------------------------
#if defined(__gfx1250__) && __has_builtin(__builtin_amdgcn_global_load_async_to_lds_b128)
#define ASYNC_CP 1
#else
#define ASYNC_CP 0
#endif

__device__ __forceinline__ void cp16_async(unsigned short* lds, const unsigned short* g) {
#if ASYNC_CP
    __builtin_amdgcn_global_load_async_to_lds_b128(
        (__attribute__((address_space(1))) v4i*)(unsigned short*)g,
        (__attribute__((address_space(3))) v4i*)lds, 0, 0);
#else
    *(U4*)lds = *(const U4*)g;
#endif
}
__device__ __forceinline__ void async_join() {
#if ASYNC_CP
    asm volatile("s_wait_asynccnt 0x0" ::: "memory");
#endif
}

// ---------------------------------------------------------------------------
// GEMM: Y[m,n] = sum_k W[m,k] * X[k,n] + bias[m]     (M=N=K=1024, batch = grid.z)
//   BTRANS=1 : X is fp32 (K,N) row-major (transpose-stage into LDS, cvt bf16)
//   BTRANS=0 : X is bf16 ushort (N,K) row-major (async-copy: already B^T)
//   OUT_MODE 0: fp32 (M,N)      1: bf16 (M,N)      2: bf16 transposed (b,h,t,d)
// Block: 256 thr = 8 waves, 128x64 tile; wave = 32x32 (4 WMMAs / K-step).
// ---------------------------------------------------------------------------
template<int OUT_MODE, bool BTRANS>
__global__ __launch_bounds__(256) void gemm128_wmma(
    const float* __restrict__ W, const float* __restrict__ Xf,
    const unsigned short* __restrict__ Xb, const float* __restrict__ bias,
    float* __restrict__ Yf, unsigned short* __restrict__ Yb)
{
    constexpr int KD = 1024, ND = 1024;
    __shared__ unsigned short As[128 * 32];  // (m,k) bf16
    __shared__ unsigned short Bs[64 * 32];   // B^T: (n,k) bf16

    const int tid  = threadIdx.x;
    const int lane = tid & 31, wave = tid >> 5;
    const int wr = wave >> 1, wc = wave & 1;
    const int half = lane >> 4, lr = lane & 15;
    const int m0 = blockIdx.y * 128, n0 = blockIdx.x * 64;
    const int z  = blockIdx.z;

    f32x8 a00, a01, a10, a11;
#pragma unroll
    for (int i = 0; i < 8; ++i) { a00[i] = 0.f; a01[i] = 0.f; a10[i] = 0.f; a11[i] = 0.f; }

    const int arow = tid >> 1;         // 0..127
    const int acol = (tid & 1) * 16;   // 0/16
    const int brow = tid >> 3;         // 0..31  (BTRANS scatter)
    const int bcol = (tid & 7) * 8;    // 0..56
    const int crow = tid >> 2;         // 0..63  (direct async copy)
    const int ccol = (tid & 3) * 8;    // 0..24

    for (int kk = 0; kk < KD; kk += 32) {
        {   // stage A (convert fp32 -> bf16): 16 elements / thread
            const float* ap = W + (size_t)(m0 + arow) * KD + kk + acol;
            if (kk + 32 < KD) __builtin_prefetch(ap + 32, 0, 1);
            float4 f0 = *(const float4*)ap;
            float4 f1 = *(const float4*)(ap + 4);
            float4 f2 = *(const float4*)(ap + 8);
            float4 f3 = *(const float4*)(ap + 12);
            unsigned short* d = &As[arow * 32 + acol];
            d[0]  = f2bf(f0.x); d[1]  = f2bf(f0.y); d[2]  = f2bf(f0.z); d[3]  = f2bf(f0.w);
            d[4]  = f2bf(f1.x); d[5]  = f2bf(f1.y); d[6]  = f2bf(f1.z); d[7]  = f2bf(f1.w);
            d[8]  = f2bf(f2.x); d[9]  = f2bf(f2.y); d[10] = f2bf(f2.z); d[11] = f2bf(f2.w);
            d[12] = f2bf(f3.x); d[13] = f2bf(f3.y); d[14] = f2bf(f3.z); d[15] = f2bf(f3.w);
        }
        if constexpr (BTRANS) {        // stage B^T with fp32->bf16 transpose
            const float* xp = Xf + (size_t)z * KD * ND + (size_t)(kk + brow) * ND + n0 + bcol;
            float4 f0 = *(const float4*)xp;
            float4 f1 = *(const float4*)(xp + 4);
            Bs[(bcol + 0) * 32 + brow] = f2bf(f0.x);
            Bs[(bcol + 1) * 32 + brow] = f2bf(f0.y);
            Bs[(bcol + 2) * 32 + brow] = f2bf(f0.z);
            Bs[(bcol + 3) * 32 + brow] = f2bf(f0.w);
            Bs[(bcol + 4) * 32 + brow] = f2bf(f1.x);
            Bs[(bcol + 5) * 32 + brow] = f2bf(f1.y);
            Bs[(bcol + 6) * 32 + brow] = f2bf(f1.z);
            Bs[(bcol + 7) * 32 + brow] = f2bf(f1.w);
        } else {                       // B already bf16 (n,k): async global->LDS
            cp16_async(&Bs[crow * 32 + ccol],
                       Xb + (size_t)z * ND * KD + (size_t)(n0 + crow) * KD + kk + ccol);
            async_join();
        }
        __syncthreads();

        const unsigned short* ap0 = &As[(wr * 32 + lr) * 32];
        const unsigned short* ap1 = &As[(wr * 32 + 16 + lr) * 32];
        bf16x16 fa0 = frag_ld(ap0 + half * 16, ap0 + half * 16 + 8);
        bf16x16 fa1 = frag_ld(ap1 + half * 16, ap1 + half * 16 + 8);
        const unsigned short* bp0 = &Bs[(wc * 32 + lr) * 32 + half * 16];
        const unsigned short* bp1 = &Bs[(wc * 32 + 16 + lr) * 32 + half * 16];
        bf16x16 fb0 = frag_ld(bp0, bp0 + 8);
        bf16x16 fb1 = frag_ld(bp1, bp1 + 8);
        a00 = WMMA_BF16(fa0, fb0, a00);
        a01 = WMMA_BF16(fa0, fb1, a01);
        a10 = WMMA_BF16(fa1, fb0, a10);
        a11 = WMMA_BF16(fa1, fb1, a11);
        __syncthreads();
    }

    const int mb0 = m0 + wr * 32 + half * 8;      // rows of a00/a01
    const int mb1 = mb0 + 16;                     // rows of a10/a11
    const int c0  = n0 + wc * 32 + lr;
    const int c1  = c0 + 16;

    if constexpr (OUT_MODE == 0) {
        float* y = Yf + (size_t)z * 1024 * ND;
#pragma unroll
        for (int i = 0; i < 8; ++i) {
            float b0 = bias[mb0 + i], b1 = bias[mb1 + i];
            y[(size_t)(mb0 + i) * ND + c0] = a00[i] + b0;
            y[(size_t)(mb0 + i) * ND + c1] = a01[i] + b0;
            y[(size_t)(mb1 + i) * ND + c0] = a10[i] + b1;
            y[(size_t)(mb1 + i) * ND + c1] = a11[i] + b1;
        }
    } else if constexpr (OUT_MODE == 1) {
        unsigned short* y = Yb + (size_t)z * 1024 * ND;
#pragma unroll
        for (int i = 0; i < 8; ++i) {
            float b0 = bias[mb0 + i], b1 = bias[mb1 + i];
            y[(size_t)(mb0 + i) * ND + c0] = f2bf(a00[i] + b0);
            y[(size_t)(mb0 + i) * ND + c1] = f2bf(a01[i] + b0);
            y[(size_t)(mb1 + i) * ND + c0] = f2bf(a10[i] + b1);
            y[(size_t)(mb1 + i) * ND + c1] = f2bf(a11[i] + b1);
        }
    } else {   // transposed per-head layout (B,H,T,64): 8 consecutive d per lane
        const int h0 = mb0 >> 6, d0 = mb0 & 63;
        const int h1 = mb1 >> 6, d1 = mb1 & 63;
        unsigned short t00[8], t01[8], t10[8], t11[8];
#pragma unroll
        for (int i = 0; i < 8; ++i) {
            float b0 = bias[mb0 + i], b1 = bias[mb1 + i];
            t00[i] = f2bf(a00[i] + b0);
            t01[i] = f2bf(a01[i] + b0);
            t10[i] = f2bf(a10[i] + b1);
            t11[i] = f2bf(a11[i] + b1);
        }
        *(U4*)&Yb[(((size_t)(z * 16 + h0)) * 1024 + c0) * 64 + d0] = pack8(t00);
        *(U4*)&Yb[(((size_t)(z * 16 + h0)) * 1024 + c1) * 64 + d0] = pack8(t01);
        *(U4*)&Yb[(((size_t)(z * 16 + h1)) * 1024 + c0) * 64 + d1] = pack8(t10);
        *(U4*)&Yb[(((size_t)(z * 16 + h1)) * 1024 + c1) * 64 + d1] = pack8(t11);
    }
}

// ---------------------------------------------------------------------------
// Flash attention with +-4 relative-position band.
//   qT,kT: (B,H,T,64) bf16    vN: (B,C,T) bf16 (= V^T per head)
//   aOut : (B,T,C) bf16  (directly consumable as B^T by the Wo GEMM)
// Grid: (T/128, B*H). Block 256 = 8 waves; wave owns 16 query rows, full d=64.
// ---------------------------------------------------------------------------
__global__ __launch_bounds__(256) void attn_wmma(
    const unsigned short* __restrict__ qT, const unsigned short* __restrict__ kT,
    const unsigned short* __restrict__ vN,
    const float* __restrict__ erk, const float* __restrict__ erv,
    unsigned short* __restrict__ aOut)
{
    constexpr int T = 1024;
    __shared__ unsigned short Qs[128 * 64];     // (tq_local, d)
    __shared__ unsigned short Ks[64 * 64];      // (tk_local, d)  == B^T for QK^T
    __shared__ unsigned short Vs[64 * 64];      // (d, tk_local)  == B^T for P.V
    __shared__ unsigned short Ps[8][16 * 64];   // per-wave exp(S) tile, bf16
    __shared__ float rbs[128][12];              // q . rel_k[d4], raw dot
    __shared__ float rkL[9 * 64];
    __shared__ float rvL[9 * 64];

    const int tid  = threadIdx.x;
    const int lane = tid & 31, wave = tid >> 5;
    const int half = lane >> 4, lr = lane & 15;
    const int bh = blockIdx.y, b = bh >> 4, h = bh & 15;
    const int q0 = blockIdx.x * 128;
    const int qw = wave * 16;

    {   // stage Q block: 128x64 bf16 = 1024 x b128 (async -> LDS)
        const unsigned short* src = qT + ((size_t)bh * T + q0) * 64;
        for (int it = tid; it < 1024; it += 256)
            cp16_async(&Qs[it * 8], src + it * 8);
    }
    for (int it = tid; it < 576; it += 256) { rkL[it] = erk[it]; rvL[it] = erv[it]; }
    async_join();
    __syncthreads();

    // rel-k band logits: rbs[r][d4] = dot(Q[r], emb_rel_k[d4]) (unscaled)
    for (int task = tid; task < 128 * 9; task += 256) {
        int r = task / 9, dd = task - r * 9;
        const unsigned short* qp = &Qs[r * 64];
        const float* kp = &rkL[dd * 64];
        float s = 0.f;
#pragma unroll 8
        for (int d = 0; d < 64; ++d) s += bf2f(qp[d]) * kp[d];
        rbs[r][dd] = s;
    }
    __syncthreads();

    f32x8 O0, O1, O2, O3;
    float mrow[8], lrow[8];
#pragma unroll
    for (int i = 0; i < 8; ++i) {
        O0[i] = 0.f; O1[i] = 0.f; O2[i] = 0.f; O3[i] = 0.f;
        mrow[i] = -1e30f; lrow[i] = 0.f;
    }

    const unsigned short* kbase = kT + (size_t)bh * T * 64;
    const unsigned short* vbase = vN + ((size_t)(b * 1024 + h * 64)) * T;
    const float scale = 0.125f;                 // 1/sqrt(64)
    const int rloc = qw + half * 8;             // row-in-block base (+i)

    for (int kb = 0; kb < T; kb += 64) {
        {   // stage K block (tk-major, coalesced, async)
            const unsigned short* src = kbase + (size_t)kb * 64;
            for (int it = tid; it < 512; it += 256)
                cp16_async(&Ks[it * 8], src + it * 8);
        }
        {   // stage V^T block (d-major rows, coalesced along tk, async)
            for (int it = tid; it < 512; it += 256) {
                int row = it >> 3, ch = it & 7;
                cp16_async(&Vs[it * 8], vbase + (size_t)row * T + kb + ch * 8);
            }
        }
        async_join();
        __syncthreads();

        // ---- S = Q K^T (4 tiles of 16x16, K-dim = 64 -> 2 wmma each) ----
        f32x8 s0, s1, s2, s3;
#pragma unroll
        for (int i = 0; i < 8; ++i) { s0[i] = 0.f; s1[i] = 0.f; s2[i] = 0.f; s3[i] = 0.f; }
#pragma unroll
        for (int dk = 0; dk < 2; ++dk) {
            const unsigned short* ap = &Qs[(qw + lr) * 64 + dk * 32];
            bf16x16 fa = frag_ld(ap + half * 16, ap + half * 16 + 8);
            const unsigned short* b0 = &Ks[(lr) * 64 + dk * 32 + half * 16];
            const unsigned short* b1 = &Ks[(16 + lr) * 64 + dk * 32 + half * 16];
            const unsigned short* b2 = &Ks[(32 + lr) * 64 + dk * 32 + half * 16];
            const unsigned short* b3 = &Ks[(48 + lr) * 64 + dk * 32 + half * 16];
            s0 = WMMA_BF16(fa, frag_ld(b0, b0 + 8), s0);
            s1 = WMMA_BF16(fa, frag_ld(b1, b1 + 8), s1);
            s2 = WMMA_BF16(fa, frag_ld(b2, b2 + 8), s2);
            s3 = WMMA_BF16(fa, frag_ld(b3, b3 + 8), s3);
        }

        // ---- scale + relative-key band term (|tk - tq| <= 4) ----
#pragma unroll
        for (int i = 0; i < 8; ++i) {
            int tq = q0 + rloc + i;
            int d4 = kb + lr - tq + 4;     // tile 0; tiles 1..3 are +16 each
            const float* rb = rbs[rloc + i];
            s0[i] = s0[i] * scale + (((unsigned)(d4)      < 9u) ? scale * rb[d4]      : 0.f);
            s1[i] = s1[i] * scale + (((unsigned)(d4 + 16) < 9u) ? scale * rb[d4 + 16] : 0.f);
            s2[i] = s2[i] * scale + (((unsigned)(d4 + 32) < 9u) ? scale * rb[d4 + 32] : 0.f);
            s3[i] = s3[i] * scale + (((unsigned)(d4 + 48) < 9u) ? scale * rb[d4 + 48] : 0.f);
        }

        // ---- online softmax (row stats replicated across each 16-lane half) ----
#pragma unroll
        for (int i = 0; i < 8; ++i) {
            float v = fmaxf(fmaxf(s0[i], s1[i]), fmaxf(s2[i], s3[i]));
            v = fmaxf(v, __shfl_xor(v, 1));
            v = fmaxf(v, __shfl_xor(v, 2));
            v = fmaxf(v, __shfl_xor(v, 4));
            v = fmaxf(v, __shfl_xor(v, 8));
            float mn = fmaxf(mrow[i], v);
            float alpha = __expf(mrow[i] - mn);
            mrow[i] = mn;
            lrow[i] *= alpha;
            O0[i] *= alpha; O1[i] *= alpha; O2[i] *= alpha; O3[i] *= alpha;
            s0[i] = __expf(s0[i] - mn);
            s1[i] = __expf(s1[i] - mn);
            s2[i] = __expf(s2[i] - mn);
            s3[i] = __expf(s3[i] - mn);
            float rs = s0[i] + s1[i] + s2[i] + s3[i];
            rs += __shfl_xor(rs, 1);
            rs += __shfl_xor(rs, 2);
            rs += __shfl_xor(rs, 4);
            rs += __shfl_xor(rs, 8);
            lrow[i] += rs;
        }

        // ---- stage P (bf16) to LDS: C-layout -> A-layout round trip ----
        unsigned short* pw = Ps[wave];
#pragma unroll
        for (int i = 0; i < 8; ++i) {
            int r = half * 8 + i;
            pw[r * 64 +      lr] = f2bf(s0[i]);
            pw[r * 64 + 16 + lr] = f2bf(s1[i]);
            pw[r * 64 + 32 + lr] = f2bf(s2[i]);
            pw[r * 64 + 48 + lr] = f2bf(s3[i]);
        }

        // ---- relative-value band: O[q] += P[q, q+delta] * rel_v[delta] ----
        if (kb <= q0 + qw + 19 && kb + 63 >= q0 + qw - 4) {
#pragma unroll
            for (int i = 0; i < 8; ++i) {
                int tq = q0 + rloc + i;
#pragma unroll
                for (int d4 = 0; d4 < 9; ++d4) {
                    int loc = tq + d4 - 4 - kb;
                    if ((unsigned)loc < 64u) {
                        float p = bf2f(pw[(half * 8 + i) * 64 + loc]);
                        O0[i] += p * rvL[d4 * 64 +      lr];
                        O1[i] += p * rvL[d4 * 64 + 16 + lr];
                        O2[i] += p * rvL[d4 * 64 + 32 + lr];
                        O3[i] += p * rvL[d4 * 64 + 48 + lr];
                    }
                }
            }
        }

        // ---- O += P V (4 d-tiles, K-dim = 64 -> 2 wmma each) ----
#pragma unroll
        for (int kt = 0; kt < 2; ++kt) {
            const unsigned short* pp = &pw[lr * 64 + kt * 32];
            bf16x16 fp = frag_ld(pp + half * 16, pp + half * 16 + 8);
            const unsigned short* v0 = &Vs[(lr) * 64 + kt * 32 + half * 16];
            const unsigned short* v1 = &Vs[(16 + lr) * 64 + kt * 32 + half * 16];
            const unsigned short* v2 = &Vs[(32 + lr) * 64 + kt * 32 + half * 16];
            const unsigned short* v3 = &Vs[(48 + lr) * 64 + kt * 32 + half * 16];
            O0 = WMMA_BF16(fp, frag_ld(v0, v0 + 8), O0);
            O1 = WMMA_BF16(fp, frag_ld(v1, v1 + 8), O1);
            O2 = WMMA_BF16(fp, frag_ld(v2, v2 + 8), O2);
            O3 = WMMA_BF16(fp, frag_ld(v3, v3 + 8), O3);
        }
        __syncthreads();
    }

    // ---- normalize + store (B,T,C) bf16 ----
#pragma unroll
    for (int i = 0; i < 8; ++i) {
        float inv = 1.0f / lrow[i];
        int tq = q0 + qw + half * 8 + i;
        unsigned short* yp = aOut + ((size_t)b * T + tq) * 1024 + h * 64;
        yp[lr]      = f2bf(O0[i] * inv);
        yp[16 + lr] = f2bf(O1[i] * inv);
        yp[32 + lr] = f2bf(O2[i] * inv);
        yp[48 + lr] = f2bf(O3[i] * inv);
    }
}

// ---------------------------------------------------------------------------
extern "C" void kernel_launch(void* const* d_in, const int* in_sizes, int n_in,
                              void* d_out, int out_size, void* d_ws, size_t ws_size,
                              hipStream_t stream) {
    const float* x   = (const float*)d_in[0];
    const float* kv  = (const float*)d_in[1];
    const float* Wq  = (const float*)d_in[2];
    const float* bq  = (const float*)d_in[3];
    const float* Wk  = (const float*)d_in[4];
    const float* bk  = (const float*)d_in[5];
    const float* Wv  = (const float*)d_in[6];
    const float* bv  = (const float*)d_in[7];
    const float* Wo  = (const float*)d_in[8];
    const float* bo  = (const float*)d_in[9];
    const float* erk = (const float*)d_in[10];
    const float* erv = (const float*)d_in[11];

    unsigned short* ws = (unsigned short*)d_ws;
    const size_t SZ = (size_t)2 * 1024 * 1024;   // elements per bf16 buffer (4 MB)
    unsigned short* qT  = ws;                    // (B,H,T,64)
    unsigned short* kTp = ws + SZ;               // (B,H,T,64)
    unsigned short* vNb = ws + 2 * SZ;           // (B,C,T)
    unsigned short* aO  = ws + 3 * SZ;           // (B,T,C)

    dim3 blk(256);
    dim3 gP(16, 8, 2);                           // (N/64, M/128, batch)

    gemm128_wmma<2, true ><<<gP, blk, 0, stream>>>(Wq, x,  nullptr, bq, nullptr, qT);
    gemm128_wmma<2, true ><<<gP, blk, 0, stream>>>(Wk, kv, nullptr, bk, nullptr, kTp);
    gemm128_wmma<1, true ><<<gP, blk, 0, stream>>>(Wv, kv, nullptr, bv, nullptr, vNb);

    attn_wmma<<<dim3(8, 32), blk, 0, stream>>>(qT, kTp, vNb, erk, erv, aO);

    gemm128_wmma<0, false><<<gP, blk, 0, stream>>>(Wo, nullptr, aO, bo, (float*)d_out, nullptr);
}
